// SemanticConsistencyLoss_21311627722971
// MI455X (gfx1250) — compile-verified, matrix operand order checked by reading, not run on hardware
//
#include <hip/hip_runtime.h>

#define NPIX   (512 * 512)
#define CF     256
#define CPAD   160
#define NT     10        // class tiles of 16 (covers 0..159, labels < 150)
#define PIXB   1024      // pixels per block (K-chunk)
#define KSTEP  32

typedef __attribute__((ext_vector_type(16))) __bf16 v16bf;
typedef __attribute__((ext_vector_type(8)))  float  v8f;
typedef __attribute__((ext_vector_type(4)))  float  v4f;
typedef __attribute__((ext_vector_type(4)))  int    v4i;

union BF16x16 {
    v16bf v;
    unsigned short h[16];
    v4i q[2];
};

__device__ __forceinline__ unsigned short bf16_rne(float f) {
    unsigned int u = __float_as_uint(f);
    u += 0x7FFFu + ((u >> 16) & 1u);   // round to nearest even
    return (unsigned short)(u >> 16);
}

// ---------------------------------------------------------------------------
// One-hot segment-sum as WMMA GEMM: Sums[cls, ch] += OneHot^T x Feat^T
// block = 512 threads = 16 waves; wave w owns channel tile w (16 channels).
// Waves 0..NT-1 additionally build the shared one-hot B tiles in LDS.
// ---------------------------------------------------------------------------
__global__ void __launch_bounds__(512)
center_gemm_kernel(const float* __restrict__ feat,
                   const int* __restrict__ labels,
                   float* __restrict__ sums) {
    __shared__ v4i bTiles[2][NT][64];   // double-buffered: 2 * 10 * 1KB = 20KB

    const int tid  = threadIdx.x;
    const int wave = tid >> 5;          // 0..15 -> channel tile
    const int lane = tid & 31;
    const int hlf  = lane >> 4;         // lane half selects K sub-range
    const int m    = lane & 15;         // row (A: channel) / col (B: class)
    const int k0   = hlf * 8;           // K offset for this lane half

    const int p0 = blockIdx.x * PIXB;
    const float* frow = feat + (size_t)(wave * 16 + m) * (size_t)NPIX;

    v8f acc[NT] = {};

    for (int step = 0; step < PIXB / KSTEP; ++step) {
        const int p   = p0 + step * KSTEP;
        const int buf = step & 1;

        // ---- A tile: 16 f32 per lane, layout matches 16-bit A VGPR map ----
        // elems 0..7  -> pixels p+k0+0..7 ; elems 8..15 -> pixels p+k0+16..23
        const v4f* fp = (const v4f*)(frow + p + k0);
        v4f a0 = fp[0];
        v4f a1 = fp[1];
        v4f a2 = fp[4];   // +16 floats
        v4f a3 = fp[5];

        // ---- build one-hot B tiles into LDS (waves 0..NT-1) ----
        if (wave < NT) {
            const int c = wave * 16 + m;            // class column this lane owns
            const v4i* lp = (const v4i*)(labels + p + k0);
            v4i L0 = lp[0], L1 = lp[1], L2 = lp[4], L3 = lp[5];
            int lab[16] = {L0[0], L0[1], L0[2], L0[3],
                           L1[0], L1[1], L1[2], L1[3],
                           L2[0], L2[1], L2[2], L2[3],
                           L3[0], L3[1], L3[2], L3[3]};
            BF16x16 b;
#pragma unroll
            for (int e = 0; e < 16; ++e)
                b.h[e] = (lab[e] == c) ? (unsigned short)0x3F80u
                                       : (unsigned short)0u;
            bTiles[buf][wave][lane * 2 + 0] = b.q[0];
            bTiles[buf][wave][lane * 2 + 1] = b.q[1];
        }

        // ---- f32 -> bf16 hi/lo split (exact against one-hot weights) ----
        float av[16] = {a0[0], a0[1], a0[2], a0[3],
                        a1[0], a1[1], a1[2], a1[3],
                        a2[0], a2[1], a2[2], a2[3],
                        a3[0], a3[1], a3[2], a3[3]};
        BF16x16 ahi, alo;
#pragma unroll
        for (int e = 0; e < 16; ++e) {
            unsigned short h = bf16_rne(av[e]);
            ahi.h[e] = h;
            float hf = __uint_as_float(((unsigned int)h) << 16);
            alo.h[e] = bf16_rne(av[e] - hf);
        }

        __syncthreads();   // B tiles for this buffer are ready

#pragma unroll
        for (int t = 0; t < NT; ++t) {
            BF16x16 b;
            b.q[0] = bTiles[buf][t][lane * 2 + 0];
            b.q[1] = bTiles[buf][t][lane * 2 + 1];
            acc[t] = __builtin_amdgcn_wmma_f32_16x16x32_bf16(
                false, ahi.v, false, b.v, (short)0, acc[t], false, false);
            acc[t] = __builtin_amdgcn_wmma_f32_16x16x32_bf16(
                false, alo.v, false, b.v, (short)0, acc[t], false, false);
        }
    }

    // ---- drain accumulators: D layout lane m = class col, VGPR r + 8*hlf = ch row
#pragma unroll
    for (int t = 0; t < NT; ++t) {
#pragma unroll
        for (int r = 0; r < 8; ++r) {
            int cls = t * 16 + m;
            int chn = wave * 16 + r + hlf * 8;
            atomicAdd(&sums[cls * CF + chn], acc[t][r]);
        }
    }
}

// ---------------------------------------------------------------------------
__global__ void __launch_bounds__(256)
hist_kernel(const int* __restrict__ labels, unsigned int* __restrict__ cnt) {
    __shared__ unsigned int h[CPAD];
    const int tid = threadIdx.x;
    if (tid < CPAD) h[tid] = 0u;
    __syncthreads();
    const int base = blockIdx.x * 256 * 16;
#pragma unroll
    for (int i = 0; i < 16; ++i) {
        int lab = labels[base + i * 256 + tid];
        atomicAdd(&h[lab], 1u);
    }
    __syncthreads();
    if (tid < CPAD) atomicAdd(&cnt[tid], h[tid]);
}

__global__ void zero_kernel(unsigned int* __restrict__ p, int n) {
    int i = blockIdx.x * 256 + threadIdx.x;
    if (i < n) p[i] = 0u;
}

__global__ void __launch_bounds__(256)
finalize_kernel(const float* __restrict__ sums,
                const unsigned int* __restrict__ cnts,
                float* __restrict__ out) {
    __shared__ float red[256];
    const float* s0 = sums;
    const float* s1 = sums + CPAD * CF;
    const unsigned int* c0 = cnts;
    const unsigned int* c1 = cnts + CPAD;

    float acc = 0.f;
    for (int i = threadIdx.x; i < 150 * CF; i += 256) {
        int c = i >> 8;                       // CF == 256
        float a = s0[i] / ((float)c0[c] + 1e-8f);
        float b = s1[i] / ((float)c1[c] + 1e-8f);
        float d = a - b;
        acc += d * d;
    }
    red[threadIdx.x] = acc;
    __syncthreads();
    for (int s = 128; s > 0; s >>= 1) {
        if (threadIdx.x < s) red[threadIdx.x] += red[threadIdx.x + s];
        __syncthreads();
    }
    if (threadIdx.x == 0) out[0] = sqrtf(red[0]);
}

// ---------------------------------------------------------------------------
extern "C" void kernel_launch(void* const* d_in, const int* in_sizes, int n_in,
                              void* d_out, int out_size, void* d_ws, size_t ws_size,
                              hipStream_t stream) {
    const float* src_fea = (const float*)d_in[0];
    const float* trg_fea = (const float*)d_in[1];
    const int*   src_lab = (const int*)d_in[2];
    const int*   trg_lab = (const int*)d_in[3];
    float* out = (float*)d_out;

    float* sums = (float*)d_ws;                                  // [2][CPAD][CF]
    unsigned int* cnts = (unsigned int*)(sums + 2 * CPAD * CF);  // [2][CPAD]

    const int zn = 2 * CPAD * CF + 2 * CPAD;
    zero_kernel<<<(zn + 255) / 256, 256, 0, stream>>>((unsigned int*)d_ws, zn);

    hist_kernel<<<NPIX / (256 * 16), 256, 0, stream>>>(src_lab, cnts);
    hist_kernel<<<NPIX / (256 * 16), 256, 0, stream>>>(trg_lab, cnts + CPAD);

    center_gemm_kernel<<<NPIX / PIXB, 512, 0, stream>>>(src_fea, src_lab, sums);
    center_gemm_kernel<<<NPIX / PIXB, 512, 0, stream>>>(trg_fea, trg_lab,
                                                        sums + CPAD * CF);

    finalize_kernel<<<1, 256, 0, stream>>>(sums, cnts, out);
}
